// MultiHeadAttention_56324201120428
// MI455X (gfx1250) — compile-verified, matrix operand order checked by reading, not run on hardware
//
#include <hip/hip_runtime.h>

#define D_MODEL 1024
#define SEQ     2048
#define NHEAD   16
#define DK      64
#define BATCH   4

typedef __attribute__((ext_vector_type(16))) __bf16  v16bf;
typedef __attribute__((ext_vector_type(8)))  __bf16  v8bf;
typedef __attribute__((ext_vector_type(8)))  float   v8f;

union FragU { v16bf v; v8bf h[2]; };

// Per-lane base offset for a 16x32 A-fragment / 32x16 B-fragment (bf16, WMMA
// 16x16x32 layout, ISA 7.12.2): lane<16 -> row (lane&15), K in {0..7,16..23};
// lane>=16 -> K in {8..15,24..31}. Caller adds k0.
__device__ __forceinline__ size_t frag_off(int ld) {
  int lane = threadIdx.x & 31;
  return (size_t)(lane & 15) * ld + (size_t)((lane >> 4) << 3);
}

__device__ __forceinline__ v16bf frag_ld_bf16(const __bf16* p) {
  FragU f;
  f.h[0] = *reinterpret_cast<const v8bf*>(p);
  f.h[1] = *reinterpret_cast<const v8bf*>(p + 16);
  return f.v;
}

__device__ __forceinline__ v8f wmma_bf16(v16bf a, v16bf b, v8f c) {
  return __builtin_amdgcn_wmma_f32_16x16x32_bf16(false, a, false, b, (short)0, c,
                                                 false, false);
}

// ---------------------------------------------------------------------------
// fp32 -> bf16 one-pass conversion (native v_cvt, packed 8 elems/thread)
// ---------------------------------------------------------------------------
__global__ void cvt_kernel(const float* __restrict__ s, __bf16* __restrict__ d) {
  const size_t i = ((size_t)blockIdx.x * 256 + threadIdx.x) * 8;
  float4 a = *reinterpret_cast<const float4*>(s + i);
  float4 b = *reinterpret_cast<const float4*>(s + i + 4);
  v8bf o;
  o[0] = (__bf16)a.x; o[1] = (__bf16)a.y; o[2] = (__bf16)a.z; o[3] = (__bf16)a.w;
  o[4] = (__bf16)b.x; o[5] = (__bf16)b.y; o[6] = (__bf16)b.z; o[7] = (__bf16)b.w;
  *reinterpret_cast<v8bf*>(d + i) = o;
}

// ---------------------------------------------------------------------------
// Projection GEMM (bf16 in, f32 accum): dst = bf16((X @ W^T + bias) * scale)
// X: (8192,1024) bf16 row-major; W: (1024,1024) bf16 row-major (out,in).
// Wave tile 32M x 64N: 2 A-frags x 4 B-frags -> 8 WMMAs per K-chunk of 32.
// vtrans==0: dst layout (b,h,s,dk)  (Q and K)
// vtrans==1: dst layout (b,h,dk,s)  (V transposed for the P@V WMMA)
// ---------------------------------------------------------------------------
__global__ void proj_kernel(const __bf16* __restrict__ X, const __bf16* __restrict__ W,
                            const float* __restrict__ bias, __bf16* __restrict__ dst,
                            float scale, int vtrans) {
  const int w    = threadIdx.x >> 5;
  const int lane = threadIdx.x & 31;
  const int c    = lane & 15;
  const int rh   = (lane >> 4) << 3;
  const int m0   = blockIdx.y * 64 + (w >> 2) * 32;
  const int n0   = blockIdx.x * 256 + (w & 3) * 64;
  const size_t fo = frag_off(D_MODEL);

  v8f vz = {0.f, 0.f, 0.f, 0.f, 0.f, 0.f, 0.f, 0.f};
  v8f acc[2][4];
#pragma unroll
  for (int i = 0; i < 2; ++i)
#pragma unroll
    for (int j = 0; j < 4; ++j) acc[i][j] = vz;

  const __bf16* Abase = X + (size_t)m0 * D_MODEL + fo;
  for (int k0 = 0; k0 < D_MODEL; k0 += 32) {
    v16bf a0 = frag_ld_bf16(Abase + k0);
    v16bf a1 = frag_ld_bf16(Abase + (size_t)16 * D_MODEL + k0);
#pragma unroll
    for (int j = 0; j < 4; ++j) {
      v16bf wf = frag_ld_bf16(W + (size_t)(n0 + j * 16) * D_MODEL + fo + k0);
      acc[0][j] = wmma_bf16(a0, wf, acc[0][j]);
      acc[1][j] = wmma_bf16(a1, wf, acc[1][j]);
    }
  }

#pragma unroll
  for (int i = 0; i < 2; ++i) {
#pragma unroll
    for (int j = 0; j < 4; ++j) {
      const int n  = n0 + j * 16 + c;
      const int hh = n >> 6, dh = n & 63;
      const float bn = bias[n];
#pragma unroll
      for (int r = 0; r < 8; ++r) {
        const int m  = m0 + i * 16 + r + rh;
        const int bb = m >> 11, s = m & (SEQ - 1);
        const float val = (acc[i][j][r] + bn) * scale;
        const size_t idx = vtrans
            ? ((size_t)((bb * NHEAD + hh) * DK + dh) * SEQ + s)
            : ((size_t)((bb * NHEAD + hh) * SEQ + s) * DK + dh);
        dst[idx] = (__bf16)val;
      }
    }
  }
}

// ---------------------------------------------------------------------------
// Flash attention: one wave owns 16 query rows, streams K/V in T-chunks of 32.
// Q was pre-scaled by log2(e)/sqrt(D) at projection => softmax uses exp2.
// P is bounced through LDS to convert accumulator layout -> A-fragment layout.
// ---------------------------------------------------------------------------
__global__ void attn_kernel(const __bf16* __restrict__ Qh,
                            const __bf16* __restrict__ Kh,
                            const __bf16* __restrict__ Vt,
                            __bf16* __restrict__ Obf) {
  __shared__ __bf16 Pt[8][512];   // per-wave 16x32 bf16 P tile

  const int w    = threadIdx.x >> 5;
  const int lane = threadIdx.x & 31;
  const int c    = lane & 15;
  const int rh   = (lane >> 4) << 3;
  const int h    = blockIdx.y;
  const int b    = blockIdx.z;
  const int bh   = b * NHEAD + h;
  const int q0   = (blockIdx.x * 8 + w) * 16;

  const size_t fo64 = frag_off(DK);
  const size_t foS  = frag_off(SEQ);
  const size_t foL  = frag_off(32);

  const __bf16* Qbase = Qh + ((size_t)bh * SEQ + q0) * DK;
  const v16bf qa0 = frag_ld_bf16(Qbase + fo64);        // d 0..31
  const v16bf qa1 = frag_ld_bf16(Qbase + fo64 + 32);   // d 32..63

  const __bf16* Kbase = Kh + (size_t)bh * SEQ * DK;    // (s, dk)
  const __bf16* Vbase = Vt + (size_t)bh * DK * SEQ;    // (dk, s)

  float m_r[8], l_r[8];
  v8f vz = {0.f, 0.f, 0.f, 0.f, 0.f, 0.f, 0.f, 0.f};
  v8f oacc[4];
#pragma unroll
  for (int r = 0; r < 8; ++r) { m_r[r] = -1e30f; l_r[r] = 0.f; }
#pragma unroll
  for (int j = 0; j < 4; ++j) oacc[j] = vz;

  for (int t0 = 0; t0 < SEQ; t0 += 32) {
    // scores: 16 x 32 chunk (two 16x16 accumulators)
    v8f s_lo = vz, s_hi = vz;
    s_lo = wmma_bf16(qa0, frag_ld_bf16(Kbase + (size_t)t0 * DK + fo64), s_lo);
    s_lo = wmma_bf16(qa1, frag_ld_bf16(Kbase + (size_t)t0 * DK + fo64 + 32), s_lo);
    s_hi = wmma_bf16(qa0, frag_ld_bf16(Kbase + (size_t)(t0 + 16) * DK + fo64), s_hi);
    s_hi = wmma_bf16(qa1, frag_ld_bf16(Kbase + (size_t)(t0 + 16) * DK + fo64 + 32), s_hi);

    // online softmax update, per accumulator row
#pragma unroll
    for (int r = 0; r < 8; ++r) {
      float mv = fmaxf(s_lo[r], s_hi[r]);
      mv = fmaxf(mv, __shfl_xor(mv, 1, 32));
      mv = fmaxf(mv, __shfl_xor(mv, 2, 32));
      mv = fmaxf(mv, __shfl_xor(mv, 4, 32));
      mv = fmaxf(mv, __shfl_xor(mv, 8, 32));
      const float mnew  = fmaxf(m_r[r], mv);
      const float alpha = __builtin_amdgcn_exp2f(m_r[r] - mnew);
      const float pl    = __builtin_amdgcn_exp2f(s_lo[r] - mnew);
      const float ph    = __builtin_amdgcn_exp2f(s_hi[r] - mnew);
      m_r[r] = mnew;
      float rs = pl + ph;
      rs += __shfl_xor(rs, 1, 32);
      rs += __shfl_xor(rs, 2, 32);
      rs += __shfl_xor(rs, 4, 32);
      rs += __shfl_xor(rs, 8, 32);
      l_r[r] = l_r[r] * alpha + rs;
#pragma unroll
      for (int j = 0; j < 4; ++j) oacc[j][r] *= alpha;
      Pt[w][(r + rh) * 32 + c]      = (__bf16)pl;
      Pt[w][(r + rh) * 32 + 16 + c] = (__bf16)ph;
    }

    // P (16x32) @ V-chunk (32x64): B-fragments contiguous thanks to Vt layout
    const v16bf pa = frag_ld_bf16(&Pt[w][0] + foL);
#pragma unroll
    for (int j = 0; j < 4; ++j) {
      v16bf vb = frag_ld_bf16(Vbase + (size_t)(j * 16) * SEQ + foS + t0);
      oacc[j] = wmma_bf16(pa, vb, oacc[j]);
    }
  }

  // normalize and store merged-head bf16 (b, s, d)
  __bf16* obase = Obf + ((size_t)b * SEQ + q0) * D_MODEL + h * DK;
#pragma unroll
  for (int r = 0; r < 8; ++r) {
    const float inv = 1.0f / l_r[r];
    const size_t rb = (size_t)(r + rh) * D_MODEL;
#pragma unroll
    for (int j = 0; j < 4; ++j)
      obase[rb + j * 16 + c] = (__bf16)(oacc[j][r] * inv);
  }
}

// ---------------------------------------------------------------------------
// Output projection: Y = Obf @ Wo^T + bo + residual  (fp32 out, into d_out)
// Wave tile 32M x 64N, all-bf16 operands.
// ---------------------------------------------------------------------------
__global__ void outproj_kernel(const __bf16* __restrict__ A,
                               const __bf16* __restrict__ W,
                               const float* __restrict__ bias,
                               const float* __restrict__ resid,
                               float* __restrict__ Y) {
  const int w    = threadIdx.x >> 5;
  const int lane = threadIdx.x & 31;
  const int c    = lane & 15;
  const int rh   = (lane >> 4) << 3;
  const int m0   = blockIdx.y * 64 + (w >> 2) * 32;
  const int n0   = blockIdx.x * 256 + (w & 3) * 64;
  const size_t fo = frag_off(D_MODEL);

  v8f vz = {0.f, 0.f, 0.f, 0.f, 0.f, 0.f, 0.f, 0.f};
  v8f acc[2][4];
#pragma unroll
  for (int i = 0; i < 2; ++i)
#pragma unroll
    for (int j = 0; j < 4; ++j) acc[i][j] = vz;

  const __bf16* Abase = A + (size_t)m0 * D_MODEL + fo;
  for (int k0 = 0; k0 < D_MODEL; k0 += 32) {
    v16bf a0 = frag_ld_bf16(Abase + k0);
    v16bf a1 = frag_ld_bf16(Abase + (size_t)16 * D_MODEL + k0);
#pragma unroll
    for (int j = 0; j < 4; ++j) {
      v16bf wf = frag_ld_bf16(W + (size_t)(n0 + j * 16) * D_MODEL + fo + k0);
      acc[0][j] = wmma_bf16(a0, wf, acc[0][j]);
      acc[1][j] = wmma_bf16(a1, wf, acc[1][j]);
    }
  }

#pragma unroll
  for (int i = 0; i < 2; ++i) {
#pragma unroll
    for (int j = 0; j < 4; ++j) {
      const int n = n0 + j * 16 + c;
      const float bn = bias[n];
#pragma unroll
      for (int r = 0; r < 8; ++r) {
        const int m = m0 + i * 16 + r + rh;
        const size_t idx = (size_t)m * D_MODEL + n;
        Y[idx] = acc[i][j][r] + bn + resid[idx];
      }
    }
  }
}

// ---------------------------------------------------------------------------
// LayerNorm, in place on d_out: one wave per 1024-element row.
// ---------------------------------------------------------------------------
__global__ void ln_kernel(float* __restrict__ Y, const float* __restrict__ g,
                          const float* __restrict__ be) {
  const int w    = threadIdx.x >> 5;
  const int lane = threadIdx.x & 31;
  float* p = Y + ((size_t)blockIdx.x * 8 + w) * D_MODEL + lane * 4;

  float4 x[8];
  float sum = 0.f, sq = 0.f;
#pragma unroll
  for (int i = 0; i < 8; ++i) {
    x[i] = *reinterpret_cast<const float4*>(p + i * 128);
    sum += x[i].x + x[i].y + x[i].z + x[i].w;
    sq  += x[i].x * x[i].x + x[i].y * x[i].y + x[i].z * x[i].z + x[i].w * x[i].w;
  }
#pragma unroll
  for (int off = 16; off >= 1; off >>= 1) {
    sum += __shfl_xor(sum, off, 32);
    sq  += __shfl_xor(sq, off, 32);
  }
  const float mean = sum * (1.0f / 1024.0f);
  const float inv  = rsqrtf(sq * (1.0f / 1024.0f) - mean * mean + 1e-5f);
#pragma unroll
  for (int i = 0; i < 8; ++i) {
    const int cc = i * 128 + lane * 4;
    float4 o;
    o.x = (x[i].x - mean) * inv * g[cc + 0] + be[cc + 0];
    o.y = (x[i].y - mean) * inv * g[cc + 1] + be[cc + 1];
    o.z = (x[i].z - mean) * inv * g[cc + 2] + be[cc + 2];
    o.w = (x[i].w - mean) * inv * g[cc + 3] + be[cc + 3];
    *reinterpret_cast<float4*>(p + i * 128) = o;
  }
}

extern "C" void kernel_launch(void* const* d_in, const int* in_sizes, int n_in,
                              void* d_out, int out_size, void* d_ws, size_t ws_size,
                              hipStream_t stream) {
  const float* q  = (const float*)d_in[0];
  const float* k  = (const float*)d_in[1];
  const float* v  = (const float*)d_in[2];
  const float* Wq = (const float*)d_in[3];
  const float* bq = (const float*)d_in[4];
  const float* Wk = (const float*)d_in[5];
  const float* bk = (const float*)d_in[6];
  const float* Wv = (const float*)d_in[7];
  const float* bv = (const float*)d_in[8];
  const float* Wo = (const float*)d_in[9];
  const float* bo = (const float*)d_in[10];
  const float* gamma = (const float*)d_in[11];
  const float* beta  = (const float*)d_in[12];
  float* out = (float*)d_out;

  const size_t NE = (size_t)BATCH * SEQ * D_MODEL;  // 8M elements
  const size_t NW = (size_t)D_MODEL * D_MODEL;      // 1M elements
  __bf16* Qh  = (__bf16*)d_ws;                      // (b,h,s,dk), Q pre-scaled
  __bf16* Kh  = Qh + NE;                            // (b,h,s,dk)
  __bf16* Vt  = Kh + NE;                            // (b,h,dk,s)
  __bf16* Obf = Vt + NE;                            // (b,s,d)
  __bf16* qb  = Obf + NE;                           // bf16 copies of inputs
  __bf16* kb  = qb + NE;
  __bf16* vb  = kb + NE;
  __bf16* Wqb = vb + NE;                            // bf16 copies of weights
  __bf16* Wkb = Wqb + NW;
  __bf16* Wvb = Wkb + NW;
  __bf16* Wob = Wvb + NW;

  dim3 blk(256);
  const int cvtN = (int)(NE / 2048);                // 4096 blocks for 8M elems
  const int cvtW = (int)(NW / 2048);                // 512 blocks for 1M elems

  cvt_kernel<<<cvtN, blk, 0, stream>>>(q, qb);
  cvt_kernel<<<cvtN, blk, 0, stream>>>(k, kb);
  cvt_kernel<<<cvtN, blk, 0, stream>>>(v, vb);
  cvt_kernel<<<cvtW, blk, 0, stream>>>(Wq, Wqb);
  cvt_kernel<<<cvtW, blk, 0, stream>>>(Wk, Wkb);
  cvt_kernel<<<cvtW, blk, 0, stream>>>(Wv, Wvb);
  cvt_kernel<<<cvtW, blk, 0, stream>>>(Wo, Wob);

  dim3 gg(D_MODEL / 256, (BATCH * SEQ) / 64);       // (4, 128)

  // fold 1/sqrt(D_MODEL) and log2(e) into Q so softmax can use v_exp_f32
  const float qscale = 1.44269504f / 32.0f;

  proj_kernel<<<gg, blk, 0, stream>>>(qb, Wqb, bq, Qh, qscale, 0);
  proj_kernel<<<gg, blk, 0, stream>>>(kb, Wkb, bk, Kh, 1.0f, 0);
  proj_kernel<<<gg, blk, 0, stream>>>(vb, Wvb, bv, Vt, 1.0f, 1);
  attn_kernel<<<dim3(SEQ / 16 / 8, NHEAD, BATCH), blk, 0, stream>>>(Qh, Kh, Vt, Obf);
  outproj_kernel<<<gg, blk, 0, stream>>>(Obf, Wob, bo, q, out);
  ln_kernel<<<dim3((BATCH * SEQ) / 8), blk, 0, stream>>>(out, gamma, beta);
}
